// EncodeContext_31164282700383
// MI455X (gfx1250) — compile-verified, matrix operand order checked by reading, not run on hardware
//
#include <hip/hip_runtime.h>

// ---------------------------------------------------------------------------
// CDNA5 (gfx1250) LSTM, bf16 WMMA + Tensor Data Mover staging.
// z = [x_t | h_{t-1}] (M=1024 x K=1024) @ Wt^T (K=1024 x N=2048) + bias
// Wt is pre-transposed/converted to bf16 and panel-reordered so each
// step-block's 128 z-rows form one contiguous [128][1024] panel.
// K-loop is peeled into: x-phase (VMEM A + TDM B), h-phase (TDM A + TDM B),
// epilogue.  Double-buffered LDS, one barrier per iteration.
// ---------------------------------------------------------------------------

typedef __attribute__((ext_vector_type(16))) __bf16 v16bf;
typedef __attribute__((ext_vector_type(8)))  float  v8f;

#define BATCH 1024
#define TSTEPS 64
#define DIM 512           // D == H == 512
#define KDIM 1024         // D + H
#define NZ 2048           // 4 * H
#define BM 64             // block tile M (batch rows)
#define BN 128            // block tile N (4 gates x 32 hidden)
#define NH 32             // hidden units per block
#define KT 32             // K step per WMMA tile
#define NKT (KDIM / KT)   // 32 K iterations
#define PITCH 40          // LDS row pitch in bf16 elems (32 + 8 pad) = 80 B
#define ZPITCH 132        // LDS z row pitch in f32

// ---- Tensor Data Mover availability (device pass only) --------------------
#if defined(__gfx1250__) && __has_builtin(__builtin_amdgcn_tensor_load_to_lds)
#define USE_TDM 1
#if __has_include(<hip/amd_detail/amd_gfx1250_TDM.h>)
#define TDM_6ARG 1
#endif
#else
#define USE_TDM 0
#endif

#if USE_TDM
typedef __attribute__((ext_vector_type(4))) unsigned tdm_u32x4;
typedef __attribute__((ext_vector_type(8))) int      tdm_i32x8;
typedef __attribute__((ext_vector_type(4))) int      tdm_i32x4;

// Load a [rows] x [32 elems] bf16 tile (row stride = strideElems) into LDS,
// padding each 64-byte row with 16 bytes (pad_interval 3 -> 16 dwords,
// pad_amount 3 -> 4 dwords) => 80 B LDS pitch matching PITCH.
__device__ __forceinline__ void tdm_load_tile(const unsigned short* gsrc,
                                              unsigned ldsOff,
                                              unsigned rows,
                                              unsigned strideElems) {
    unsigned long long ga = (unsigned long long)(const void*)gsrc;
    tdm_u32x4 g0;
    g0.x = 1u;                                           // count=1 (valid)
    g0.y = ldsOff;                                       // lds_addr (bytes)
    g0.z = (unsigned)ga;                                 // global_addr[31:0]
    g0.w = (unsigned)((ga >> 32) & 0x01FFFFFFull)        // global_addr[56:32]
         | (2u << 30);                                   // type=2 ("image")
    tdm_i32x8 g1 = {
        (int)((1u << 16)              // data_size = 1 -> 2 bytes
            | (1u << 20)              // pad_enable
            | (3u << 22)              // pad_interval: 16 dwords
            | (3u << 25)),            // pad_amount:    4 dwords
        (int)(strideElems << 16),     // tensor_dim0 (bits 63:48)
        (int)(4096u << 16),           // tensor_dim1 (bits 95:80), generous
        (int)(32u << 16),             // tile_dim0 = 32   (bits 127:112)
        (int)rows,                    // tile_dim1        (bits 143:128)
        (int)strideElems,             // tensor_dim0_stride (bits 191:160)
        0, 0};
    tdm_i32x4 gz = {0, 0, 0, 0};
#if defined(TDM_6ARG)
    tdm_i32x8 gz8 = {0, 0, 0, 0, 0, 0, 0, 0};
    __builtin_amdgcn_tensor_load_to_lds(g0, g1, gz, gz, gz8, 0);
#else
    __builtin_amdgcn_tensor_load_to_lds(g0, g1, gz, gz, 0);
#endif
}

__device__ __forceinline__ void wait_tensorcnt0() {
#if __has_builtin(__builtin_amdgcn_s_wait_tensorcnt)
    __builtin_amdgcn_s_wait_tensorcnt(0);
#else
    asm volatile("s_wait_tensorcnt 0x0" ::: "memory");
#endif
}
#endif  // USE_TDM

union FragCast {
    uint4 u[2];
    v16bf v;
};

__device__ __forceinline__ unsigned short f2bf(float f) {
    union { float f; unsigned u; } cv;
    cv.f = f;
    unsigned r = cv.u + 0x7FFFu + ((cv.u >> 16) & 1u);  // round-to-nearest-even
    return (unsigned short)(r >> 16);
}

__device__ __forceinline__ float fast_sigmoid(float x) {
    return 1.0f / (1.0f + __expf(-x));
}

__device__ __forceinline__ float fast_tanh(float x) {
    x = fminf(fmaxf(x, -15.0f), 15.0f);
    float e = __expf(2.0f * x);
    return (e - 1.0f) / (e + 1.0f);
}

__device__ __forceinline__ unsigned lds_off32(const void* p) {
    return (unsigned)(unsigned long long)p;  // low 32 bits = LDS byte offset
}

// 16x32 bf16 fragment row-chunk: lane<16 -> K {0..7,16..23},
// lane>=16 -> K {8..15,24..31}  (ISA 7.12.2 16-bit A/B layout).
__device__ __forceinline__ v16bf load_frag(const unsigned short* row, int half) {
    const uint4* p = (const uint4*)row;
    FragCast f;
    f.u[0] = p[half];
    f.u[1] = p[2 + half];
    return f.v;
}

// ---------------------------------------------------------------------------
// One-time: Wt[r][k] = bf16( k < 512 ? W[k][n] : U[k-512][n] ) where the row
// index r packs n = gate*512 + nh into contiguous per-block panels:
//   r = (nh/32)*128 + gate*32 + (nh%32)
// ---------------------------------------------------------------------------
__global__ __launch_bounds__(256)
void transpose_weights(const float* __restrict__ W, const float* __restrict__ U,
                       unsigned short* __restrict__ Wt) {
    __shared__ float tile[32][33];
    const int bk = blockIdx.x * 32;
    const int bn = blockIdx.y * 32;
    const int tx = threadIdx.x;   // 0..31
    const int ty = threadIdx.y;   // 0..7
    for (int i = ty; i < 32; i += 8) {
        const int k = bk + i;
        const int n = bn + tx;
        float v = (k < DIM) ? W[(size_t)k * NZ + n] : U[(size_t)(k - DIM) * NZ + n];
        tile[i][tx] = v;
    }
    __syncthreads();
    for (int i = ty; i < 32; i += 8) {
        const int n = bn + i;
        const int gate = n >> 9;
        const int nh = n & (DIM - 1);
        const int r = (nh >> 5) * BN + gate * NH + (nh & (NH - 1));
        Wt[(size_t)r * KDIM + bk + tx] = f2bf(tile[tx][i]);
    }
}

// ---------------------------------------------------------------------------
__global__ void init_state(unsigned short* __restrict__ h0, float* __restrict__ c) {
    const int i = blockIdx.x * 256 + threadIdx.x;
    if (i < BATCH * DIM) {
        h0[i] = 0;
        c[i] = 0.0f;
    }
}

// ---------------------------------------------------------------------------
// One LSTM time step.  grid (16 m-tiles, 16 nh-tiles), 256 threads (8 waves).
// ---------------------------------------------------------------------------
__global__ __launch_bounds__(256)
void lstm_step(const float* __restrict__ x,            // [B][T][D]
               const unsigned short* __restrict__ Wt,  // panels [16][128][1024] bf16
               const float* __restrict__ bias,         // [NZ]
               const unsigned short* __restrict__ h_in,  // [B][H] bf16
               unsigned short* __restrict__ h_out,       // [B][H] bf16
               float* __restrict__ c,                    // [B][H]
               float* __restrict__ out,                  // [B][T][H]
               int t) {
    union __align__(16) Smem {
        struct {
            unsigned short a[2][BM * PITCH];   // A tiles  64 x 32, ping/pong
            unsigned short b[2][BN * PITCH];   // B tiles 128 x 32, ping/pong
        } ab;
        float z[BM * ZPITCH];                  // z tile   64 x 128
    };
    __shared__ Smem sm;

    const int tid  = threadIdx.x;
    const int lane = tid & 31;
    const int wave = tid >> 5;
    const int half = lane >> 4;
    const int l15  = lane & 15;
    const int m0   = blockIdx.x * BM;    // batch-row base
    const int nh0  = blockIdx.y * NH;    // hidden-unit base

    const int wmBase = (wave & 1) * 32;  // wave M offset
    const int wnBase = (wave >> 1) * 32; // wave N offset (z columns)

    // A loader: 64 rows x 32 k, 8 elems/thread
    const int ra  = tid >> 2;
    const int kha = (tid & 3) * 8;
    const float* xrow = x + (size_t)(m0 + ra) * (TSTEPS * DIM) + (size_t)t * DIM;
    const unsigned short* hbase = h_in + (size_t)m0 * DIM;   // block h panel

    // Contiguous weight panel for this block
    const unsigned short* panel = Wt + (size_t)blockIdx.y * BN * KDIM;
#if !USE_TDM
    const int rb  = tid >> 1;            // B loader fallback
    const int khb = (tid & 1) * 16;
    const unsigned short* hrow = hbase + (size_t)ra * DIM;
#endif

    v8f acc[2][2] = {};

    auto compute_tile = [&](int buf) {
        v16bf af0 = load_frag(&sm.ab.a[buf][(wmBase + l15) * PITCH], half);
        v16bf af1 = load_frag(&sm.ab.a[buf][(wmBase + 16 + l15) * PITCH], half);
        v16bf bf0 = load_frag(&sm.ab.b[buf][(wnBase + l15) * PITCH], half);
        v16bf bf1 = load_frag(&sm.ab.b[buf][(wnBase + 16 + l15) * PITCH], half);
        acc[0][0] = __builtin_amdgcn_wmma_f32_16x16x32_bf16(
            false, af0, false, bf0, (short)0, acc[0][0], false, false);
        acc[0][1] = __builtin_amdgcn_wmma_f32_16x16x32_bf16(
            false, af0, false, bf1, (short)0, acc[0][1], false, false);
        acc[1][0] = __builtin_amdgcn_wmma_f32_16x16x32_bf16(
            false, af1, false, bf0, (short)0, acc[1][0], false, false);
        acc[1][1] = __builtin_amdgcn_wmma_f32_16x16x32_bf16(
            false, af1, false, bf1, (short)0, acc[1][1], false, false);
    };

    auto stage_x = [&](int buf, int kt) {   // load+convert x tile kt, return packed bf16
        const float4* xp = (const float4*)(xrow + kt * KT + kha);
        float4 v0 = xp[0];
        float4 v1 = xp[1];
        __align__(16) unsigned short tmp[8];
        tmp[0] = f2bf(v0.x); tmp[1] = f2bf(v0.y);
        tmp[2] = f2bf(v0.z); tmp[3] = f2bf(v0.w);
        tmp[4] = f2bf(v1.x); tmp[5] = f2bf(v1.y);
        tmp[6] = f2bf(v1.z); tmp[7] = f2bf(v1.w);
        return *(const uint4*)tmp;
    };
    (void)stage_x;

    // ---------------- prologue: stage tile 0 ----------------
    {
        const float4* xp = (const float4*)(xrow + kha);
        float4 v0 = xp[0];
        float4 v1 = xp[1];
        __align__(16) unsigned short tmp[8];
        tmp[0] = f2bf(v0.x); tmp[1] = f2bf(v0.y);
        tmp[2] = f2bf(v0.z); tmp[3] = f2bf(v0.w);
        tmp[4] = f2bf(v1.x); tmp[5] = f2bf(v1.y);
        tmp[6] = f2bf(v1.z); tmp[7] = f2bf(v1.w);
        *(uint4*)&sm.ab.a[0][ra * PITCH + kha] = *(const uint4*)tmp;
#if USE_TDM
        if (wave == 0) {
            tdm_load_tile(panel, lds_off32(&sm.ab.b[0][0]), BN, KDIM);
            wait_tensorcnt0();
        }
#else
        const uint4* wp = (const uint4*)(panel + (size_t)rb * KDIM + khb);
        uint4* dst = (uint4*)&sm.ab.b[0][rb * PITCH + khb];
        dst[0] = wp[0];
        dst[1] = wp[1];
#endif
    }
    __syncthreads();

    // -------- segment 1: kt = 0..14, prefetch x tile kt+1 (VMEM + TDM-B) ----
    #pragma unroll 2
    for (int kt = 0; kt < 15; ++kt) {
        const int buf = kt & 1;
        const int ktn = kt + 1;
        const float4* xp = (const float4*)(xrow + ktn * KT + kha);
        float4 v0 = xp[0];
        float4 v1 = xp[1];
#if USE_TDM
        if (wave == 0)
            tdm_load_tile(panel + ktn * KT, lds_off32(&sm.ab.b[1 - buf][0]), BN, KDIM);
#else
        const uint4* wp = (const uint4*)(panel + (size_t)rb * KDIM + ktn * KT + khb);
        uint4 br0 = wp[0];
        uint4 br1 = wp[1];
#endif
        compute_tile(buf);

        __align__(16) unsigned short tmp[8];
        tmp[0] = f2bf(v0.x); tmp[1] = f2bf(v0.y);
        tmp[2] = f2bf(v0.z); tmp[3] = f2bf(v0.w);
        tmp[4] = f2bf(v1.x); tmp[5] = f2bf(v1.y);
        tmp[6] = f2bf(v1.z); tmp[7] = f2bf(v1.w);
        *(uint4*)&sm.ab.a[1 - buf][ra * PITCH + kha] = *(const uint4*)tmp;
#if USE_TDM
        if (wave == 0) wait_tensorcnt0();
#else
        uint4* dst = (uint4*)&sm.ab.b[1 - buf][rb * PITCH + khb];
        dst[0] = br0;
        dst[1] = br1;
#endif
        __syncthreads();
    }

    // -------- segment 2: kt = 15..30, prefetch h tile kt+1 (all TDM) --------
    #pragma unroll 2
    for (int kt = 15; kt < 31; ++kt) {
        const int buf = kt & 1;
        const int ktn = kt + 1;
#if USE_TDM
        if (wave == 0) {
            tdm_load_tile(panel + ktn * KT, lds_off32(&sm.ab.b[1 - buf][0]), BN, KDIM);
            tdm_load_tile(hbase + (ktn - 16) * KT, lds_off32(&sm.ab.a[1 - buf][0]),
                          BM, DIM);
        }
        compute_tile(buf);
        if (wave == 0) wait_tensorcnt0();
#else
        const uint4 hr = *(const uint4*)(hrow + (ktn - 16) * KT + kha);
        const uint4* wp = (const uint4*)(panel + (size_t)rb * KDIM + ktn * KT + khb);
        uint4 br0 = wp[0];
        uint4 br1 = wp[1];
        compute_tile(buf);
        *(uint4*)&sm.ab.a[1 - buf][ra * PITCH + kha] = hr;
        uint4* dst = (uint4*)&sm.ab.b[1 - buf][rb * PITCH + khb];
        dst[0] = br0;
        dst[1] = br1;
#endif
        __syncthreads();
    }

    // -------- epilogue: compute tile 31 (buf = 1), no staging ---------------
    compute_tile(1);
    __syncthreads();   // all frag reads done before z overlays the LDS tiles

    // ---- park z in LDS ----
    #pragma unroll
    for (int i = 0; i < 2; ++i)
        #pragma unroll
        for (int j = 0; j < 2; ++j)
            #pragma unroll
            for (int r = 0; r < 8; ++r) {
                const int m = wmBase + i * 16 + half * 8 + r;
                const int n = wnBase + j * 16 + l15;
                sm.z[m * ZPITCH + n] = acc[i][j][r];
            }
    __syncthreads();

    // ---- gates + state update: 64 rows x 32 hidden, 8 elems/thread ----
    const int m  = tid >> 2;
    const int j0 = (tid & 3) * 8;
    const int gm = m0 + m;
    #pragma unroll
    for (int jj = 0; jj < 8; ++jj) {
        const int j = j0 + jj;
        const float zi = sm.z[m * ZPITCH + j]      + bias[nh0 + j];
        const float zf = sm.z[m * ZPITCH + 32 + j] + bias[DIM + nh0 + j];
        const float zg = sm.z[m * ZPITCH + 64 + j] + bias[2 * DIM + nh0 + j];
        const float zo = sm.z[m * ZPITCH + 96 + j] + bias[3 * DIM + nh0 + j];
        const float ig = fast_sigmoid(zi);
        const float fg = fast_sigmoid(zf);
        const float gg = fast_tanh(zg);
        const float og = fast_sigmoid(zo);
        const size_t cidx = (size_t)gm * DIM + nh0 + j;
        const float cn = fg * c[cidx] + ig * gg;
        c[cidx] = cn;
        const float h = og * fast_tanh(cn);
        out[(size_t)gm * (TSTEPS * DIM) + (size_t)t * DIM + nh0 + j] = h;
        h_out[cidx] = f2bf(h);
    }
}

// ---------------------------------------------------------------------------
extern "C" void kernel_launch(void* const* d_in, const int* in_sizes, int n_in,
                              void* d_out, int out_size, void* d_ws, size_t ws_size,
                              hipStream_t stream) {
    const float* x    = (const float*)d_in[0];  // [1024][64][512]
    const float* W    = (const float*)d_in[1];  // [512][2048]
    const float* U    = (const float*)d_in[2];  // [512][2048]
    const float* bias = (const float*)d_in[3];  // [2048]
    float* out = (float*)d_out;                 // [1024][64][512]

    char* ws = (char*)d_ws;
    unsigned short* Wt = (unsigned short*)ws;                          // 4 MB
    unsigned short* h0 = (unsigned short*)(ws + (size_t)(4u << 20));   // 1 MB
    unsigned short* h1 = (unsigned short*)(ws + (size_t)(5u << 20));   // 1 MB
    float*          c  = (float*)(ws + (size_t)(6u << 20));            // 2 MB

    transpose_weights<<<dim3(32, 64), dim3(32, 8), 0, stream>>>(W, U, Wt);
    init_state<<<(BATCH * DIM + 255) / 256, 256, 0, stream>>>(h0, c);

    unsigned short* hb[2] = {h0, h1};
    for (int t = 0; t < TSTEPS; ++t) {
        lstm_step<<<dim3(BATCH / BM, DIM / NH), 256, 0, stream>>>(
            x, Wt, bias, hb[t & 1], hb[(t + 1) & 1], c, out, t);
    }
}
